// GroupAttention_80719615361202
// MI455X (gfx1250) — matrix operand order repaired and checked by reading, not compiled, not run on hardware
//
#include <hip/hip_runtime.h>
#include <hip/hip_bf16.h>

typedef __bf16 bf16_t;
typedef __attribute__((ext_vector_type(16))) __bf16 v16bf;
typedef __attribute__((ext_vector_type(8)))  float  v8f;

#define HN    8
#define KD    32
#define DV    128
#define QKV_O 192   // 2*KD + DV
#define CG    64    // DIM / H
#define NSEQ  1024
#define BATCH 16
#define DIMC  512
#define CTOT  (HN * DV)   // 1024 channels of attention output

#define QKV_BYTES ((size_t)BATCH * HN * QKV_O * NSEQ * 2)   // 50331648
#define OB_BYTES  ((size_t)BATCH * CTOT * NSEQ * 2)         // 33554432

// ---------------------------------------------------------------------------
// CDNA5 async memory->LDS copy (ASYNCcnt-tracked), per ISA §15.18 / §08.
// ---------------------------------------------------------------------------
__device__ __forceinline__ void async_copy_b128(void* lds_dst, const void* gsrc) {
    unsigned int lo = (unsigned int)(unsigned long long)lds_dst;  // LDS offset = addr[31:0]
    asm volatile("global_load_async_to_lds_b128 %0, %1, off"
                 :: "v"(lo), "v"(gsrc)
                 : "memory");
}
__device__ __forceinline__ void wait_async0() {
    asm volatile("s_wait_asynccnt 0" ::: "memory");
}

// ---------------------------------------------------------------------------
// Kernel 0: convert proj weights fp32 -> bf16
// ---------------------------------------------------------------------------
__global__ void cvt_kernel(const float* __restrict__ src, bf16_t* __restrict__ dst, int n) {
    int i = blockIdx.x * blockDim.x + threadIdx.x;
    if (i < n) dst[i] = (bf16_t)src[i];
}

// ---------------------------------------------------------------------------
// Kernel 1: grouped 1x1 conv (QKV).  Per (b,h): [192,64] @ [64,1024] + bias.
// Fully LDS-staged; scale folded into q.  Output bf16 [B,H,192,N].
// grid = B*H*(N/128), block = 256 (8 waves)
// ---------------------------------------------------------------------------
__global__ void qkv_kernel(const float* __restrict__ x,
                           const float* __restrict__ qkv_w,
                           const float* __restrict__ qkv_b,
                           bf16_t* __restrict__ qkvb) {
    int blk = blockIdx.x;
    int nb = blk & 7, h = (blk >> 3) & 7, b = blk >> 6;
    int n0 = nb * 128;

    __shared__ bf16_t Wl[QKV_O * CG];   // 24 KB
    __shared__ bf16_t Xl[CG * 128];     // 16 KB

    int tid = threadIdx.x;
    const float* wsrc = qkv_w + (size_t)h * QKV_O * CG;
    for (int i = tid; i < QKV_O * CG; i += 256) Wl[i] = (bf16_t)wsrc[i];
    const float* xsrc = x + ((size_t)b * DIMC + (size_t)h * CG) * NSEQ;
    for (int i = tid; i < CG * 128; i += 256) {
        int c = i >> 7, j = i & 127;
        Xl[i] = (bf16_t)xsrc[(size_t)c * NSEQ + n0 + j];
    }
    __syncthreads();

    int lane = tid & 31, wv = tid >> 5;
    int col = lane & 15, half = lane >> 4;
    const float scale = 0.17677669529663687f;  // 32^-0.5

    for (int mt = 0; mt < 12; ++mt) {
        v8f acc = {0.f, 0.f, 0.f, 0.f, 0.f, 0.f, 0.f, 0.f};
        #pragma unroll
        for (int k0 = 0; k0 < CG; k0 += 32) {
            v16bf a, bm;
            #pragma unroll
            for (int i = 0; i < 8; ++i) {
                int kc = half * 16 + 2 * i;
                a[2 * i]     = Wl[(mt * 16 + (lane & 15)) * CG + k0 + kc];
                a[2 * i + 1] = Wl[(mt * 16 + (lane & 15)) * CG + k0 + kc + 1];
                int kr = k0 + half * 16 + 2 * i;
                bm[2 * i]     = Xl[kr * 128 + wv * 16 + col];
                bm[2 * i + 1] = Xl[(kr + 1) * 128 + wv * 16 + col];
            }
            acc = __builtin_amdgcn_wmma_f32_16x16x32_bf16(false, a, false, bm,
                                                          (short)0, acc, false, false);
        }
        #pragma unroll
        for (int r = 0; r < 8; ++r) {
            int o = mt * 16 + r + 8 * half;
            float v = acc[r] + qkv_b[h * QKV_O + o];
            if (o < KD) v *= scale;   // fold softmax scale into q
            qkvb[(((size_t)b * HN + h) * QKV_O + o) * NSEQ + n0 + wv * 16 + col] = (bf16_t)v;
        }
    }
}

// ---------------------------------------------------------------------------
// Kernel 2: fused flash attention.  Per 32-column step: K^T staged with a
// register transpose, V staged via CDNA5 async LDS loads; S = q^T k (2 WMMA),
// online softmax (half-wave shfl), P transposed through per-wave LDS,
// O += P V^T (8 independent WMMA chains).  ReLU fused; output n-major
// bf16 [B][N][H*D] so stores are half-wave contiguous.
// grid = B*H*(N/128), block = 256 (8 waves)
// ---------------------------------------------------------------------------
__global__ void attn_kernel(const bf16_t* __restrict__ qkvb,
                            bf16_t* __restrict__ obf) {
    int blk = blockIdx.x;
    int mb = blk & 7, h = (blk >> 3) & 7, b = blk >> 6;
    int tid = threadIdx.x, lane = tid & 31, wv = tid >> 5;
    int col = lane & 15, half = lane >> 4;

    const bf16_t* qb = qkvb + ((size_t)b * HN + h) * QKV_O * NSEQ;
    const bf16_t* kb = qb + (size_t)KD * NSEQ;
    const bf16_t* vb = qb + (size_t)(2 * KD) * NSEQ;

    int m0 = mb * 128 + wv * 16;

    __shared__ unsigned short Kt[32][32];    // [n_local][c]  (transposed)  2 KB
    __shared__ unsigned short Vt[128][32];   // [d][n_local]                8 KB
    __shared__ bf16_t         Pl[8][16][32]; // per-wave P transpose        8 KB

    // q A-fragment (16 rows x K=32) -- whole key_dim, loaded once
    v16bf aq;
    int gm = m0 + (lane & 15);
    #pragma unroll
    for (int i = 0; i < 8; ++i) {
        int kc = half * 16 + 2 * i;
        aq[2 * i]     = qb[(size_t)kc * NSEQ + gm];
        aq[2 * i + 1] = qb[(size_t)(kc + 1) * NSEQ + gm];
    }

    float mrun[8], lrun[8];
    #pragma unroll
    for (int r = 0; r < 8; ++r) { mrun[r] = -1e30f; lrun[r] = 0.f; }
    v8f acc[8];
    #pragma unroll
    for (int dt = 0; dt < 8; ++dt)
        acc[dt] = (v8f){0.f, 0.f, 0.f, 0.f, 0.f, 0.f, 0.f, 0.f};

    // staging coordinates
    int kc_row = tid >> 3, knq = (tid & 7) * 4;   // K: 32 rows x 8 quads
    int vd = tid >> 1, vnh = (tid & 1) * 16;      // V: 128 rows x 2 halves

    for (int n0 = 0; n0 < NSEQ; n0 += 32) {
        __syncthreads();   // previous iteration's tile readers done
        // ---- stage V rows: async memory -> LDS (ASYNCcnt) ----
        {
            const bf16_t* src = vb + (size_t)vd * NSEQ + n0 + vnh;
            async_copy_b128(&Vt[vd][vnh],     src);
            async_copy_b128(&Vt[vd][vnh + 8], src + 8);
        }
        // ---- stage K^T (register transpose 32c x 32n -> [n][c]) ----
        {
            const bf16_t* src = kb + (size_t)kc_row * NSEQ + n0 + knq;
            unsigned int u0 = *(const unsigned int*)(src);
            unsigned int u1 = *(const unsigned int*)(src + 2);
            Kt[knq + 0][kc_row] = (unsigned short)(u0 & 0xffffu);
            Kt[knq + 1][kc_row] = (unsigned short)(u0 >> 16);
            Kt[knq + 2][kc_row] = (unsigned short)(u1 & 0xffffu);
            Kt[knq + 3][kc_row] = (unsigned short)(u1 >> 16);
        }
        if (n0 + 32 < NSEQ)
            __builtin_prefetch(kb + (size_t)kc_row * NSEQ + n0 + 32 + knq, 0, 0);
        wait_async0();
        __syncthreads();

        // ---- S = q^T k for 32 columns (two 16-wide tiles) ----
        v8f s[2];
        #pragma unroll
        for (int sub = 0; sub < 2; ++sub) {
            const bf16_t* krow = (const bf16_t*)&Kt[sub * 16 + col][0];
            v16bf bk = *(const v16bf*)(krow + half * 16);
            v8f z = {0.f, 0.f, 0.f, 0.f, 0.f, 0.f, 0.f, 0.f};
            s[sub] = __builtin_amdgcn_wmma_f32_16x16x32_bf16(false, aq, false, bk,
                                                             (short)0, z, false, false);
        }

        // ---- online softmax: rows live across 16 lanes of each half ----
        #pragma unroll
        for (int r = 0; r < 8; ++r) {
            float tmax = fmaxf(s[0][r], s[1][r]);
            #pragma unroll
            for (int off = 1; off < 16; off <<= 1)
                tmax = fmaxf(tmax, __shfl_xor(tmax, off, 32));
            float mnew = fmaxf(mrun[r], tmax);
            float cf = __expf(mrun[r] - mnew);
            float p0 = __expf(s[0][r] - mnew);
            float p1 = __expf(s[1][r] - mnew);
            float rs = p0 + p1;
            #pragma unroll
            for (int off = 1; off < 16; off <<= 1)
                rs += __shfl_xor(rs, off, 32);
            lrun[r] = lrun[r] * cf + rs;
            mrun[r] = mnew;
            s[0][r] = p0; s[1][r] = p1;
            #pragma unroll
            for (int dt = 0; dt < 8; ++dt) acc[dt][r] *= cf;
        }

        // ---- transpose P (C layout -> A layout) through per-wave LDS ----
        #pragma unroll
        for (int r = 0; r < 8; ++r) {
            Pl[wv][r + 8 * half][col]      = (bf16_t)s[0][r];
            Pl[wv][r + 8 * half][16 + col] = (bf16_t)s[1][r];
        }
        asm volatile("s_wait_dscnt 0" ::: "memory");  // wave-local LDS RAW

        v16bf ap;
        #pragma unroll
        for (int i = 0; i < 8; ++i) {
            int kc = half * 16 + 2 * i;
            ap[2 * i]     = Pl[wv][lane & 15][kc];
            ap[2 * i + 1] = Pl[wv][lane & 15][kc + 1];
        }

        // ---- O += P @ V^T  (8 independent accumulator chains) ----
        #pragma unroll
        for (int dt = 0; dt < 8; ++dt) {
            const bf16_t* vrow = (const bf16_t*)&Vt[dt * 16 + col][0];
            v16bf bv = *(const v16bf*)(vrow + half * 16);
            acc[dt] = __builtin_amdgcn_wmma_f32_16x16x32_bf16(false, ap, false, bv,
                                                              (short)0, acc[dt], false, false);
        }
    }

    // ---- normalize + ReLU + store n-major bf16 [B][N][H*D] ----
    // Lane col spans d within a tile -> 16-lane contiguous 32B store runs.
    float inv[8];
    #pragma unroll
    for (int r = 0; r < 8; ++r) inv[r] = (lrun[r] > 0.f) ? (1.f / lrun[r]) : 0.f;
    #pragma unroll
    for (int r = 0; r < 8; ++r) {
        int m = m0 + r + 8 * half;
        bf16_t* orow = obf + ((size_t)b * NSEQ + m) * CTOT + h * DV;
        #pragma unroll
        for (int dt = 0; dt < 8; ++dt) {
            float v = fmaxf(acc[dt][r] * inv[r], 0.f);
            orow[dt * 16 + col] = (bf16_t)v;
        }
    }
}

// ---------------------------------------------------------------------------
// Kernel 3: projection GEMM + BatchNorm.  Per b: [512,1024] @ [1024,1024].
// W rows LDS-resident; O tile (n-major layout) staged via async LDS loads --
// no transpose needed.  BN folded into epilogue.
// grid = B * 32 (m-tiles) * 8 (n-blocks of 128), block = 256 (8 waves)
// ---------------------------------------------------------------------------
__global__ void proj_kernel(const bf16_t* __restrict__ obf,
                            const bf16_t* __restrict__ pwb,
                            const float* __restrict__ proj_b,
                            const float* __restrict__ gamma,
                            const float* __restrict__ beta,
                            const float* __restrict__ mean,
                            const float* __restrict__ var,
                            float* __restrict__ out) {
    int blk = blockIdx.x;
    int nb = blk & 7, mt = (blk >> 3) & 31, b = blk >> 8;
    int n0 = nb * 128;

    __shared__ bf16_t         Wl[16 * 1024];   // 32 KB: 16 out rows x K=1024
    __shared__ unsigned short Ol[128][32];     //  8 KB: [n_local][k_local]

    int tid = threadIdx.x;
    const bf16_t* wsrc = pwb + (size_t)mt * 16 * 1024;
    for (int i = tid; i < 16 * 1024; i += 256) Wl[i] = wsrc[i];

    int lane = tid & 31, wv = tid >> 5, col = lane & 15, half = lane >> 4;
    int gn = n0 + wv * 16 + col;
    int sn = tid >> 1, sch = (tid & 1) * 16;   // staging: 128 n-rows x 2 chunks

    v8f acc = {0.f, 0.f, 0.f, 0.f, 0.f, 0.f, 0.f, 0.f};
    for (int k0 = 0; k0 < CTOT; k0 += 32) {
        __syncthreads();
        // stage O tile [128 n][32 k] via async LDS loads (layout matches, no transpose)
        {
            const bf16_t* src = obf + ((size_t)b * NSEQ + n0 + sn) * CTOT + k0 + sch;
            async_copy_b128(&Ol[sn][sch],     src);
            async_copy_b128(&Ol[sn][sch + 8], src + 8);
        }
        wait_async0();
        __syncthreads();

        v16bf a;
        #pragma unroll
        for (int i = 0; i < 8; ++i) {
            int kc = half * 16 + 2 * i;
            a[2 * i]     = Wl[(lane & 15) * 1024 + k0 + kc];
            a[2 * i + 1] = Wl[(lane & 15) * 1024 + k0 + kc + 1];
        }
        const bf16_t* orow = (const bf16_t*)&Ol[wv * 16 + col][0];
        v16bf bm = *(const v16bf*)(orow + half * 16);

        acc = __builtin_amdgcn_wmma_f32_16x16x32_bf16(false, a, false, bm,
                                                      (short)0, acc, false, false);
    }

    #pragma unroll
    for (int r = 0; r < 8; ++r) {
        int o = mt * 16 + r + 8 * half;
        float alpha = gamma[o] * __frsqrt_rn(var[o] + 1e-5f);
        float bias  = proj_b[o] * alpha + beta[o] - mean[o] * alpha;
        out[((size_t)b * DIMC + o) * NSEQ + gn] = acc[r] * alpha + bias;
    }
}

// ---------------------------------------------------------------------------
extern "C" void kernel_launch(void* const* d_in, const int* in_sizes, int n_in,
                              void* d_out, int out_size, void* d_ws, size_t ws_size,
                              hipStream_t stream) {
    const float* x        = (const float*)d_in[0];
    const float* qkv_w    = (const float*)d_in[1];
    const float* qkv_b    = (const float*)d_in[2];
    const float* proj_w   = (const float*)d_in[3];
    const float* proj_b   = (const float*)d_in[4];
    const float* bn_gamma = (const float*)d_in[5];
    const float* bn_beta  = (const float*)d_in[6];
    const float* bn_mean  = (const float*)d_in[7];
    const float* bn_var   = (const float*)d_in[8];
    float* out = (float*)d_out;

    char* ws = (char*)d_ws;
    bf16_t* qkvb = (bf16_t*)ws;                             // 50331648 B
    bf16_t* obf  = (bf16_t*)(ws + QKV_BYTES);               // 33554432 B  (n-major)
    bf16_t* pwb  = (bf16_t*)(ws + QKV_BYTES + OB_BYTES);    //  1048576 B

    int nw = DIMC * CTOT;  // 524288
    cvt_kernel <<<(nw + 255) / 256, 256, 0, stream>>>(proj_w, pwb, nw);
    qkv_kernel <<<BATCH * HN * (NSEQ / 128), 256, 0, stream>>>(x, qkv_w, qkv_b, qkvb);
    attn_kernel<<<BATCH * HN * (NSEQ / 128), 256, 0, stream>>>(qkvb, obf);
    proj_kernel<<<BATCH * 32 * 8, 256, 0, stream>>>(obf, pwb, proj_b, bn_gamma,
                                                    bn_beta, bn_mean, bn_var, out);
}
